// SemiDiscreteOptimalTransport_12884901888021
// MI455X (gfx1250) — matrix operand order repaired and checked by reading, not compiled, not run on hardware
//
#include <hip/hip_runtime.h>

// Problem constants (fixed by the reference).
#define NSRC 8192
#define MTGT 16384
#define DDIM 64
#define REG_P 0.05f
#define RSPLIT 4                 // target-range splits (grid.y)
#define JPB (MTGT / RSPLIT)      // targets per block = 4096
#define SUP 64                   // targets per LDS super-chunk
#define NSUP (JPB / SUP)         // super-chunks per block = 64

typedef __attribute__((ext_vector_type(16))) _Float16     v16h;
typedef __attribute__((ext_vector_type(8)))  float        v8f;
typedef __attribute__((ext_vector_type(4)))  unsigned int u32x4;
typedef __attribute__((ext_vector_type(4)))  float        f32x4;

union V16H { u32x4 u[2]; v16h h; };
union V8F  { f32x4 f[2]; v8f  v; };

// ---- workspace layout (bytes) ----
constexpr size_t OFF_SQT  = 0;                                   // [MTGT] f32 ||t_j||^2
constexpr size_t OFF_COLT = OFF_SQT  + (size_t)MTGT * 4;         // [MTGT] f32 log2 column term
constexpr size_t OFF_PT   = OFF_COLT + (size_t)MTGT * 4;         // [64] partials sum||t||^2
constexpr size_t OFF_PP   = OFF_PT   + 64 * 4;                   // [64] partials sum psi
constexpr size_t OFF_PS   = OFF_PP   + 64 * 4;                   // [32] partials sum||s||^2
constexpr size_t OFF_SC   = OFF_PS   + 32 * 4;                   // [8]  scalars: cn, scale, base
constexpr size_t OFF_PM   = OFF_SC   + 8 * 4;                    // [RSPLIT*NSRC] running max
constexpr size_t OFF_PSM  = OFF_PM   + (size_t)RSPLIT * NSRC * 4;// [RSPLIT*NSRC] running sum
constexpr size_t OFF_PO   = OFF_PSM  + (size_t)RSPLIT * NSRC * 4;// [32] lse partial sums
constexpr size_t OFF_THI  = (OFF_PO + 32 * 4 + 255) & ~(size_t)255;
constexpr size_t OFF_TLO  = OFF_THI + (size_t)MTGT * DDIM * 2;   // f16 hi/lo targets (pre-scaled)
constexpr size_t OFF_SHI  = OFF_TLO + (size_t)MTGT * DDIM * 2;   // f16 hi/lo sources
constexpr size_t OFF_SLO  = OFF_SHI + (size_t)NSRC * DDIM * 2;
// total ~6.5 MB

__device__ __forceinline__ v8f wmma16x16x32(v16h a, v16h b, v8f c) {
  return __builtin_amdgcn_wmma_f32_16x16x32_f16(false, a, false, b, (short)0, c,
                                                false, false);
}

// Raw hardware transcendentals: single v_exp_f32 / v_log_f32 (base-2), no
// range-fixup sequence. Arguments are always <= 0 here (x - running_max), and
// inputs below -126 flush toward 0 -- exactly right for logsumexp terms.
__device__ __forceinline__ float fast_exp2(float x) { return __builtin_amdgcn_exp2f(x); }
__device__ __forceinline__ float fast_log2(float x) { return __builtin_amdgcn_logf(x); }

// Async global->LDS copy, 16B per lane. Tracked by ASYNCcnt (CDNA5).
// Generic shared-pointer low 32 bits == wave-relative LDS address (ISA 10.2).
__device__ __forceinline__ void async_b128(void* lds, const void* g) {
  unsigned int laddr = (unsigned int)(size_t)lds;
  unsigned long long gaddr = (unsigned long long)(size_t)g;
  asm volatile("global_load_async_to_lds_b128 %0, %1, off"
               :: "v"(laddr), "v"(gaddr)
               : "memory");
}
__device__ __forceinline__ void wait_async_le4() {
  asm volatile("s_wait_asynccnt 0x4" ::: "memory");
}
__device__ __forceinline__ void wait_async_le0() {
  asm volatile("s_wait_asynccnt 0x0" ::: "memory");
}

// ---- Kernel 1a: target row stats ----
__global__ __launch_bounds__(256) void k_tstats(const float* __restrict__ tgt,
                                                const float* __restrict__ psi,
                                                float* __restrict__ sq_t,
                                                float* __restrict__ part_t,
                                                float* __restrict__ part_p) {
  __shared__ float r1[256], r2[256];
  const int tid = threadIdx.x;
  const int j = blockIdx.x * 256 + tid;
  const f32x4* row = (const f32x4*)(tgt + (size_t)j * DDIM);
  float ss = 0.f;
#pragma unroll
  for (int k = 0; k < DDIM / 4; ++k) {
    f32x4 v = row[k];
    ss += v.x * v.x + v.y * v.y + v.z * v.z + v.w * v.w;
  }
  sq_t[j] = ss;
  r1[tid] = ss;
  r2[tid] = psi[j];
  __syncthreads();
  for (int s = 128; s > 0; s >>= 1) {
    if (tid < s) { r1[tid] += r1[tid + s]; r2[tid] += r2[tid + s]; }
    __syncthreads();
  }
  if (tid == 0) { part_t[blockIdx.x] = r1[0]; part_p[blockIdx.x] = r2[0]; }
}

// ---- Kernel 1b: source row stats ----
__global__ __launch_bounds__(256) void k_sstats(const float* __restrict__ src,
                                                float* __restrict__ part_s) {
  __shared__ float r1[256];
  const int tid = threadIdx.x;
  const int i = blockIdx.x * 256 + tid;
  const f32x4* row = (const f32x4*)(src + (size_t)i * DDIM);
  float ss = 0.f;
#pragma unroll
  for (int k = 0; k < DDIM / 4; ++k) {
    f32x4 v = row[k];
    ss += v.x * v.x + v.y * v.y + v.z * v.z + v.w * v.w;
  }
  r1[tid] = ss;
  __syncthreads();
  for (int s = 128; s > 0; s >>= 1) {
    if (tid < s) r1[tid] += r1[tid + s];
    __syncthreads();
  }
  if (tid == 0) part_s[blockIdx.x] = r1[0];
}

// ---- Kernel 2: fold partials into scalars (deterministic serial) ----
__global__ void k_scalars(const float* __restrict__ part_t,
                          const float* __restrict__ part_p,
                          const float* __restrict__ part_s,
                          float* __restrict__ scal) {
  float st = 0.f, sp = 0.f, ssrc = 0.f;
  for (int b = 0; b < MTGT / 256; ++b) { st += part_t[b]; sp += part_p[b]; }
  for (int b = 0; b < NSRC / 256; ++b) ssrc += part_s[b];
  const float cn = st / (float)MTGT;               // cost_norm = mean ||t||^2
  const float LOG2E = 1.4426950408889634f;
  scal[0] = cn;
  scal[1] = 2.0f * LOG2E / (REG_P * cn);           // scale folded into f16 targets
  scal[2] = sp / (float)MTGT + (ssrc / (float)NSRC) / cn;  // exact row-term + mean(psi)
}

// ---- Kernel 3a: targets -> scaled f16 hi/lo + log2 column term ----
__global__ __launch_bounds__(256) void k_convt(const float* __restrict__ tgt,
                                               const float* __restrict__ psi,
                                               const float* __restrict__ sq_t,
                                               const float* __restrict__ scal,
                                               _Float16* __restrict__ thi,
                                               _Float16* __restrict__ tlo,
                                               float* __restrict__ colt2) {
  const int j = blockIdx.x * 256 + threadIdx.x;
  const float cn = scal[0];
  const float sc = scal[1];
  const float LOG2E = 1.4426950408889634f;
  const float LNM = logf((float)MTGT);
  colt2[j] = LOG2E * ((psi[j] - sq_t[j] / cn) / REG_P - LNM);
  const float* row = tgt + (size_t)j * DDIM;
  _Float16* oh = thi + (size_t)j * DDIM;
  _Float16* ol = tlo + (size_t)j * DDIM;
  for (int k = 0; k < DDIM; ++k) {
    const float u = sc * row[k];
    const _Float16 h = (_Float16)u;
    oh[k] = h;
    ol[k] = (_Float16)(u - (float)h);
  }
}

// ---- Kernel 3b: sources -> f16 hi/lo ----
__global__ __launch_bounds__(256) void k_convs(const float* __restrict__ src,
                                               _Float16* __restrict__ shi,
                                               _Float16* __restrict__ slo) {
  const int i = blockIdx.x * 256 + threadIdx.x;
  const float* row = src + (size_t)i * DDIM;
  _Float16* oh = shi + (size_t)i * DDIM;
  _Float16* ol = slo + (size_t)i * DDIM;
  for (int k = 0; k < DDIM; ++k) {
    const float u = row[k];
    const _Float16 h = (_Float16)u;
    oh[k] = h;
    ol[k] = (_Float16)(u - (float)h);
  }
}

// ---- Kernel 4: WMMA + async-LDS-staged streaming logsumexp ----
// Grid (64, RSPLIT). Block (bx,by): sources [bx*128,+128) (8 waves x 16),
// targets [by*4096,+4096) streamed in 64-row super-chunks double-buffered in LDS.
// All 8 waves share each staged slab -> 8x less L2 target traffic.
__global__ __launch_bounds__(256) void sdot_main(const _Float16* __restrict__ thi,
                                                 const _Float16* __restrict__ tlo,
                                                 const _Float16* __restrict__ shi,
                                                 const _Float16* __restrict__ slo,
                                                 const float* __restrict__ colt2,
                                                 float* __restrict__ part_m,
                                                 float* __restrict__ part_sum) {
  __shared__ __attribute__((aligned(16))) char smem[2][16384];  // [hi 8KB | lo 8KB]

  const int tid = threadIdx.x;
  const int lane = tid & 31;
  const int wv = tid >> 5;
  const int half = lane >> 4;
  const int lm = lane & 15;
  const int i0 = blockIdx.x * 128 + wv * 16;     // this wave's 16 sources
  const int jbase = blockIdx.y * JPB;            // this block's target range

  // Loop-invariant B operand: column n = source i0+lm.
  // B layout: lanes 0-15 hold K=0..15, lanes 16-31 hold K=16..31.
  const _Float16* sph = shi + (size_t)(i0 + lm) * DDIM;
  const _Float16* spl = slo + (size_t)(i0 + lm) * DDIM;
  V16H bh[2], bl[2];
#pragma unroll
  for (int q = 0; q < 2; ++q) {
    const int kb = q * 32 + half * 16;
    bh[q].u[0] = *(const u32x4*)(sph + kb);
    bh[q].u[1] = *(const u32x4*)(sph + kb + 8);
    bl[q].u[0] = *(const u32x4*)(spl + kb);
    bl[q].u[1] = *(const u32x4*)(spl + kb + 8);
  }

  const char* gth = (const char*)thi + (size_t)jbase * DDIM * 2;
  const char* gtl = (const char*)tlo + (size_t)jbase * DDIM * 2;

  // Issue async stage for super-chunks 0 and 1 (4 ASYNC ops / wave each).
#pragma unroll
  for (int p = 0; p < 2; ++p) {
    char* lb = &smem[p][0];
    const char* gh = gth + (size_t)p * SUP * 128;
    const char* gl = gtl + (size_t)p * SUP * 128;
    async_b128(lb + tid * 16,         gh + tid * 16);
    async_b128(lb + 4096 + tid * 16,  gh + 4096 + tid * 16);
    async_b128(lb + 8192 + tid * 16,  gl + tid * 16);
    async_b128(lb + 12288 + tid * 16, gl + 4096 + tid * 16);
  }

  float mrun = -__builtin_huge_valf();
  float srun = 0.0f;
  const int c0 = half * 16;  // A layout: lanes 0-15 K 0-15, lanes 16-31 K 16-31

  for (int scn = 0; scn < NSUP; ++scn) {
    // ASYNCcnt decrements in order: <=4 outstanding => this super-chunk landed.
    if (scn + 1 < NSUP) wait_async_le4(); else wait_async_le0();
    __syncthreads();  // make all waves' staged data visible

    const char* bb = &smem[scn & 1][0];
#pragma unroll
    for (int c = 0; c < 4; ++c) {
      const char* rh = bb + (size_t)(c * 16 + lm) * 128;  // A row in LDS (hi)
      V16H ah[2], al[2];
#pragma unroll
      for (int q = 0; q < 2; ++q) {
        const int o0 = (q * 32 + c0) * 2;
        ah[q].u[0] = *(const u32x4*)(rh + o0);
        ah[q].u[1] = *(const u32x4*)(rh + o0 + 16);
        al[q].u[0] = *(const u32x4*)(rh + 8192 + o0);
        al[q].u[1] = *(const u32x4*)(rh + 8192 + o0 + 16);
      }
      const int j0g = jbase + scn * SUP + c * 16;
      V8F cc;
      const f32x4* cp = (const f32x4*)(colt2 + j0g + 8 * half);
      cc.f[0] = cp[0];
      cc.f[1] = cp[1];

      v8f acc = cc.v;  // C = log2 column term; hi/lo split dot: hh + hl + lh
      acc = wmma16x16x32(ah[0].h, bh[0].h, acc);
      acc = wmma16x16x32(ah[1].h, bh[1].h, acc);
      acc = wmma16x16x32(ah[0].h, bl[0].h, acc);
      acc = wmma16x16x32(ah[1].h, bl[1].h, acc);
      acc = wmma16x16x32(al[0].h, bh[0].h, acc);
      acc = wmma16x16x32(al[1].h, bh[1].h, acc);

      float cmax = acc[0];
#pragma unroll
      for (int r = 1; r < 8; ++r) cmax = fmaxf(cmax, acc[r]);
      const float mnew = fmaxf(mrun, cmax);
      float s = srun * fast_exp2(mrun - mnew);
#pragma unroll
      for (int r = 0; r < 8; ++r) s += fast_exp2(acc[r] - mnew);
      mrun = mnew;
      srun = s;
    }

    __syncthreads();  // all waves done reading this buffer
    if (scn + 2 < NSUP) {  // refill it two super-chunks ahead
      char* lb = &smem[scn & 1][0];
      const char* gh = gth + (size_t)(scn + 2) * SUP * 128;
      const char* gl = gtl + (size_t)(scn + 2) * SUP * 128;
      async_b128(lb + tid * 16,         gh + tid * 16);
      async_b128(lb + 4096 + tid * 16,  gh + 4096 + tid * 16);
      async_b128(lb + 8192 + tid * 16,  gl + tid * 16);
      async_b128(lb + 12288 + tid * 16, gl + 4096 + tid * 16);
    }
  }

  // Merge lane halves (targets 0-7 with 8-15 of each chunk); write per-source
  // partial (max, sum) for this target range.
  const float pm = __shfl_xor(mrun, 16);
  const float ps = __shfl_xor(srun, 16);
  const float mh = fmaxf(mrun, pm);
  const float sh2 = srun * fast_exp2(mrun - mh) + ps * fast_exp2(pm - mh);
  if (lane < 16) {
    part_m[(size_t)blockIdx.y * NSRC + i0 + lm] = mh;
    part_sum[(size_t)blockIdx.y * NSRC + i0 + lm] = sh2;
  }
}

// ---- Kernel 5: merge the RSPLIT target-range partials per source (deterministic) ----
__global__ __launch_bounds__(256) void k_merge(const float* __restrict__ part_m,
                                               const float* __restrict__ part_sum,
                                               float* __restrict__ part_o) {
  __shared__ float red[256];
  const int tid = threadIdx.x;
  const int i = blockIdx.x * 256 + tid;
  float Mx = -__builtin_huge_valf(), S = 0.0f;
#pragma unroll
  for (int r = 0; r < RSPLIT; ++r) {
    const float mw = part_m[(size_t)r * NSRC + i];
    const float sw = part_sum[(size_t)r * NSRC + i];
    const float mn = fmaxf(Mx, mw);
    S = S * fast_exp2(Mx - mn) + sw * fast_exp2(mw - mn);
    Mx = mn;
  }
  red[tid] = Mx + fast_log2(S);  // log2-lse' for source i
  __syncthreads();
  for (int s = 128; s > 0; s >>= 1) {
    if (tid < s) red[tid] += red[tid + s];
    __syncthreads();
  }
  if (tid == 0) part_o[blockIdx.x] = red[0];
}

// ---- Kernel 6: final deterministic sum ----
__global__ void k_final(const float* __restrict__ part_o,
                        const float* __restrict__ scal,
                        float* __restrict__ out) {
  float t = 0.0f;
  for (int b = 0; b < NSRC / 256; ++b) t += part_o[b];
  out[0] = scal[2] + (-REG_P * 0.69314718055994531f / (float)NSRC) * t;
}

extern "C" void kernel_launch(void* const* d_in, const int* in_sizes, int n_in,
                              void* d_out, int out_size, void* d_ws, size_t ws_size,
                              hipStream_t stream) {
  (void)in_sizes; (void)n_in; (void)out_size; (void)ws_size;
  const float* src = (const float*)d_in[0];  // [NSRC, DDIM]
  const float* tgt = (const float*)d_in[1];  // [MTGT, DDIM]
  const float* psi = (const float*)d_in[2];  // [MTGT]

  char* ws = (char*)d_ws;
  float* sq_t   = (float*)(ws + OFF_SQT);
  float* colt2  = (float*)(ws + OFF_COLT);
  float* part_t = (float*)(ws + OFF_PT);
  float* part_p = (float*)(ws + OFF_PP);
  float* part_s = (float*)(ws + OFF_PS);
  float* scal   = (float*)(ws + OFF_SC);
  float* part_m = (float*)(ws + OFF_PM);
  float* part_sm= (float*)(ws + OFF_PSM);
  float* part_o = (float*)(ws + OFF_PO);
  _Float16* thi = (_Float16*)(ws + OFF_THI);
  _Float16* tlo = (_Float16*)(ws + OFF_TLO);
  _Float16* shi = (_Float16*)(ws + OFF_SHI);
  _Float16* slo = (_Float16*)(ws + OFF_SLO);

  k_tstats<<<MTGT / 256, 256, 0, stream>>>(tgt, psi, sq_t, part_t, part_p);
  k_sstats<<<NSRC / 256, 256, 0, stream>>>(src, part_s);
  k_scalars<<<1, 1, 0, stream>>>(part_t, part_p, part_s, scal);
  k_convt<<<MTGT / 256, 256, 0, stream>>>(tgt, psi, sq_t, scal, thi, tlo, colt2);
  k_convs<<<NSRC / 256, 256, 0, stream>>>(src, shi, slo);
  sdot_main<<<dim3(NSRC / 128, RSPLIT), 256, 0, stream>>>(thi, tlo, shi, slo,
                                                          colt2, part_m, part_sm);
  k_merge<<<NSRC / 256, 256, 0, stream>>>(part_m, part_sm, part_o);
  k_final<<<1, 1, 0, stream>>>(part_o, scal, (float*)d_out);
}
